// FakeQuantLinear_74517682586279
// MI455X (gfx1250) — 8-bit matrix operand order repaired and checked by reading, compiled, not run on hardware
//
#include <hip/hip_runtime.h>
#include <stdint.h>

typedef __attribute__((ext_vector_type(8))) int   v8i;
typedef __attribute__((ext_vector_type(4))) int   v4i;
typedef __attribute__((ext_vector_type(4))) float v4f;

#define QMAXF 127.0f
#define QEPS  1e-8f

__global__ void __launch_bounds__(256)
row_quant_i8_kernel(const float* __restrict__ src,
                    int8_t* __restrict__ qout,
                    float* __restrict__ scale_out,
                    int ncols  ) {
  const int lane = threadIdx.x & 31;
  const int wave = threadIdx.x >> 5;
  const int row  = blockIdx.x * 8 + wave;

  const v4f* rowp = reinterpret_cast<const v4f*>(src + (size_t)row * ncols);
  v4f v[8];
  float amax = 0.0f;
#pragma unroll
  for (int i = 0; i < 8; ++i) {
    v[i] = __builtin_nontemporal_load(rowp + i * 32 + lane);
    amax = fmaxf(amax, fabsf(v[i].x));
    amax = fmaxf(amax, fabsf(v[i].y));
    amax = fmaxf(amax, fabsf(v[i].z));
    amax = fmaxf(amax, fabsf(v[i].w));
  }
#pragma unroll
  for (int m = 16; m >= 1; m >>= 1)
    amax = fmaxf(amax, __shfl_xor(amax, m, 32));

  const float scale = fmaxf(amax * (1.0f / QMAXF), QEPS);
  const float inv   = 1.0f / scale;
  if (lane == 0) scale_out[row] = scale;

  uint32_t* qrow = reinterpret_cast<uint32_t*>(qout + (size_t)row * ncols);
#pragma unroll
  for (int i = 0; i < 8; ++i) {
    int q0 = (int)fmaxf(fminf(rintf(v[i].x * inv),  QMAXF), -QMAXF);
    int q1 = (int)fmaxf(fminf(rintf(v[i].y * inv),  QMAXF), -QMAXF);
    int q2 = (int)fmaxf(fminf(rintf(v[i].z * inv),  QMAXF), -QMAXF);
    int q3 = (int)fmaxf(fminf(rintf(v[i].w * inv),  QMAXF), -QMAXF);
    uint32_t packed = (uint32_t)(q0 & 0xff)
                    | ((uint32_t)(q1 & 0xff) << 8)
                    | ((uint32_t)(q2 & 0xff) << 16)
                    | ((uint32_t)(q3 & 0xff) << 24);
    qrow[i * 32 + lane] = packed;
  }
}

__global__ void __launch_bounds__(256, 2)
i8_wmma_gemm_kernel(const int8_t* __restrict__ qx,
                    const float* __restrict__ a_scale,
                    const int8_t* __restrict__ qw,
                    const float* __restrict__ w_scale,
                    const float* __restrict__ bias,
                    float* __restrict__ out,
                    int M, int N, int K) {
  const int lane = threadIdx.x & 31;
  const int wave = threadIdx.x >> 5;
  const int wrow = wave & 3;
  const int wcol = wave >> 2;

  const int m_base = blockIdx.y * 128 + wrow * 32;
  const int n_base = blockIdx.x * 128 + wcol * 64;

  const int lh    = lane >> 4;
  const int l15   = lane & 15;
  const int halfA = lh << 4;
  const int halfB = lh << 4;

  const int8_t* arow[2];
  arow[0] = qx + (size_t)(m_base + l15) * K;
  arow[1] = arow[0] + (size_t)16 * K;
  const int8_t* brow[4];
#pragma unroll
  for (int j = 0; j < 4; ++j)
    brow[j] = qw + (size_t)(n_base + j * 16 + l15) * K;

  v8i acc[2][4];
#pragma unroll
  for (int i = 0; i < 2; ++i)
#pragma unroll
    for (int j = 0; j < 4; ++j)
      acc[i][j] = (v8i)0;

  for (int k0 = 0; k0 < K; k0 += 64) {
    __builtin_prefetch(arow[0] + k0 + 64, 0, 3);
    __builtin_prefetch(arow[1] + k0 + 64, 0, 3);

    v8i afrag[2];
#pragma unroll
    for (int i = 0; i < 2; ++i) {
      const unsigned long long* p =
          reinterpret_cast<const unsigned long long*>(arow[i] + k0 + halfA);
      unsigned long long d0 = p[0];
      unsigned long long d1 = p[1];
      unsigned long long d2 = p[4];
      unsigned long long d3 = p[5];
      afrag[i][0] = (int)(uint32_t)d0;  afrag[i][1] = (int)(uint32_t)(d0 >> 32);
      afrag[i][2] = (int)(uint32_t)d1;  afrag[i][3] = (int)(uint32_t)(d1 >> 32);
      afrag[i][4] = (int)(uint32_t)d2;  afrag[i][5] = (int)(uint32_t)(d2 >> 32);
      afrag[i][6] = (int)(uint32_t)d3;  afrag[i][7] = (int)(uint32_t)(d3 >> 32);
    }

    v8i bfrag[4];
#pragma unroll
    for (int j = 0; j < 4; ++j) {
      const v4i* p = reinterpret_cast<const v4i*>(brow[j] + k0 + halfB);
      v4i lo = p[0];
      v4i hi = p[2];
      bfrag[j][0] = lo[0]; bfrag[j][1] = lo[1];
      bfrag[j][2] = lo[2]; bfrag[j][3] = lo[3];
      bfrag[j][4] = hi[0]; bfrag[j][5] = hi[1];
      bfrag[j][6] = hi[2]; bfrag[j][7] = hi[3];
    }

#pragma unroll
    for (int i = 0; i < 2; ++i)
#pragma unroll
      for (int j = 0; j < 4; ++j)
        acc[i][j] = __builtin_amdgcn_wmma_i32_16x16x64_iu8(
             true, afrag[i],
             true, bfrag[j],
            acc[i][j],
             false,  false);
  }

#pragma unroll
  for (int j = 0; j < 4; ++j) {
    const int n    = n_base + j * 16 + l15;
    const float ws = w_scale[n];
    const float bn = bias[n];
#pragma unroll
    for (int i = 0; i < 2; ++i) {
#pragma unroll
      for (int r = 0; r < 8; ++r) {
        const int m   = m_base + i * 16 + r + (lh << 3);
        const float s = a_scale[m] * ws;
        __builtin_nontemporal_store((float)acc[i][j][r] * s + bn,
                                    &out[(size_t)m * N + n]);
      }
    }
  }
}

extern "C" void kernel_launch(void* const* d_in, const int* in_sizes, int n_in,
                              void* d_out, int out_size, void* d_ws, size_t ws_size,
                              hipStream_t stream) {
  const float* x    = (const float*)d_in[0];
  const float* w    = (const float*)d_in[1];
  const float* bias = (const float*)d_in[2];
  float* out = (float*)d_out;

  const int N = in_sizes[2];
  const int K = in_sizes[1] / N;
  const int M = in_sizes[0] / K;

  int8_t* qx       = (int8_t*)d_ws;
  int8_t* qw       = qx + (size_t)M * K;
  float*  a_scale  = (float*)(qw + (size_t)N * K);
  float*  w_scale  = a_scale + M;

  row_quant_i8_kernel<<<M / 8, 256, 0, stream>>>(x, qx, a_scale, K);
  row_quant_i8_kernel<<<N / 8, 256, 0, stream>>>(w, qw, w_scale, K);

  dim3 grid(N / 128, M / 128);
  i8_wmma_gemm_kernel<<<grid, 256, 0, stream>>>(qx, a_scale, qw, w_scale, bias,
                                                out, M, N, K);
}
